// Decoder_44513041055824
// MI455X (gfx1250) — matrix operand order repaired and checked by reading, not compiled, not run on hardware
//
#include <hip/hip_runtime.h>
#include <hip/hip_bf16.h>

// ---------------- problem constants ----------------
#define BATCH   32
#define T_ENC   256
#define D_ENC   512
#define MEL     80
#define RR      2
#define T_DEC   500
#define PRE1    256
#define PRE2    256
#define ARNN    1024
#define DRNN    1024
#define ADIM    128
#define FILT    32
#define KSZ     31
#define NSTEPS  (T_DEC / RR)      // 250
#define GATES   (4 * ARNN)        // 4096
#define XA_K    (PRE2 + D_ENC)    // 768  (prenet | ctx)
#define DX_K    (ARNN + D_ENC)    // 1536 (attn-h | ctx)

typedef __attribute__((ext_vector_type(16))) __bf16 v16bf;
typedef __attribute__((ext_vector_type(8)))  float  v8f;
typedef __attribute__((ext_vector_type(4)))  unsigned int u32x4;
typedef __attribute__((ext_vector_type(8)))  int          i32x8;
typedef __attribute__((ext_vector_type(4)))  int          i32x4;

__device__ __forceinline__ float sigf(float x) { return 1.0f / (1.0f + __expf(-x)); }

// ---------------- TDM: DMA a contiguous bf16 panel (nDwords dwords) into LDS ----------------
// D# per CDNA5 ISA ch.8: group0 = {count, lds_addr, global_addr, type=2},
// group1 = {data_size=4B, 1-D tile: tensor_dim0=tile_dim0=nDwords, dim1=1, stride=nDwords}.
// This toolchain exposes the 6-arg builtin: (g0, g1, g2, g3, g4, cpol).
__device__ __forceinline__ void tdm_load_panel(const __bf16* gsrc, unsigned int ldsByteOff,
                                               unsigned int nDwords) {
  unsigned long long ga = (unsigned long long)(uintptr_t)gsrc;
  u32x4 g0;
  g0[0] = 1u;                                                     // count=1, no gather
  g0[1] = ldsByteOff;                                             // lds_addr [63:32]
  g0[2] = (unsigned int)(ga & 0xFFFFFFFFull);                     // global_addr[31:0]
  g0[3] = (unsigned int)((ga >> 32) & 0x1FFFFFFull) | (2u << 30); // global_addr[56:32] | type=2
  i32x8 g1;
  g1[0] = (int)(2u << 16);                                        // data_size = 4B
  g1[1] = (int)((nDwords & 0xFFFFu) << 16);                       // tensor_dim0[15:0]
  g1[2] = (int)(((nDwords >> 16) & 0xFFFFu) | (1u << 16));        // tensor_dim0[31:16] | tensor_dim1=1
  g1[3] = (int)((nDwords & 0xFFFFu) << 16);                       // tile_dim0 = nDwords (fits 16b)
  g1[4] = 1;                                                      // tile_dim1=1, tile_dim2=0
  g1[5] = (int)nDwords;                                           // tensor_dim0_stride[31:0]
  g1[6] = 0;                                                      // stride hi, dim1_stride lo
  g1[7] = 0;
  i32x4 z4 = {0, 0, 0, 0};                                        // groups 2/3: zero (2-D path)
  i32x8 z8 = {0, 0, 0, 0, 0, 0, 0, 0};
  __builtin_amdgcn_tensor_load_to_lds(g0, g1, z4, z4, z8, 0);
}

// ---------------- WMMA fragment loaders ----------------
// A: 16x32 bf16 (M x K) fragment from a row-major panel with row stride K.
// lane L: m = L&15, half = L>>4; VGPR v holds K pair base (v<4 ? 2v : 2v+8) + 8*half
__device__ __forceinline__ v16bf load_a_frag(const __bf16* __restrict__ A, int K,
                                             int m, int half, int kbase) {
  union { v16bf v; unsigned int u[8]; } f;
  const __bf16* row = A + (size_t)m * K + kbase + 8 * half;
#pragma unroll
  for (int v = 0; v < 8; ++v) {
    int kv = (v < 4) ? (2 * v) : (2 * v + 8);
    f.u[v] = *(const unsigned int*)(row + kv);
  }
  return f.v;
}

// B: 32x16 bf16 (K x N tile), B stored K-major (K rows x N cols).
// lane L owns N = n0 + (L&15); element s holds K = kbase + 8*(L>>4) + (s<8 ? s : s+8).
__device__ __forceinline__ v16bf load_b_frag(const __bf16* __restrict__ B, int N,
                                             int kbase, int lane, int n0) {
  union { v16bf v; __bf16 e[16]; } f;
  const __bf16* p = B + (size_t)(kbase + 8 * (lane >> 4)) * N + n0 + (lane & 15);
#pragma unroll
  for (int s = 0; s < 16; ++s)
    f.e[s] = p[(size_t)((s < 8) ? s : (s + 8)) * N];
  // prefetch next k-block of the (L2-resident) weight matrix
  __builtin_prefetch(p + (size_t)32 * N, 0, 1);
  return f.v;
}

// ---------------- generic bf16 WMMA GEMM: C = A1*B1 (+ A2*B2) (+ bias1 + bias2) ----------------
// All 8 waves of a block share one M-tile; the 16xK activation panel is staged
// into LDS once per segment via the Tensor Data Mover. B streams from L2.
// Requires: tilesN multiple of 8, K multiples of 32, grid = (M/16)*(tilesN/8).
__global__ void k_wmma_gemm(const __bf16* __restrict__ A1, const __bf16* __restrict__ B1, int K1,
                            const __bf16* __restrict__ A2, const __bf16* __restrict__ B2, int K2,
                            const float* __restrict__ bias1, const float* __restrict__ bias2,
                            float* __restrict__ C, int M, int N) {
  extern __shared__ __bf16 ldsA[];   // 16 x max(K1,K2) bf16
  int lane = threadIdx.x & 31;
  int wave = threadIdx.x >> 5;
  int tilesN = N >> 4;
  int blocksPerM = tilesN >> 3;      // 8 waves/block, one N-tile each
  int tM = blockIdx.x / blocksPerM;
  int tN = (blockIdx.x - tM * blocksPerM) * 8 + wave;
  int half = lane >> 4;
  int mrow = lane & 15;
  int n0 = tN * 16;

  // ---- stage A1 panel (16 x K1, contiguous) into LDS via TDM, wave 0 only ----
  if (threadIdx.x < 32) {
    tdm_load_panel(A1 + (size_t)tM * 16 * K1, (unsigned int)(uintptr_t)ldsA,
                   (unsigned int)(16 * K1) >> 1);
    __builtin_amdgcn_s_wait_tensorcnt(0);
  }
  __syncthreads();

  v8f acc = {};
  for (int k = 0; k < K1; k += 32) {
    v16bf a = load_a_frag(ldsA, K1, mrow, half, k);
    v16bf b = load_b_frag(B1, N, k, lane, n0);
    acc = __builtin_amdgcn_wmma_f32_16x16x32_bf16(false, a, false, b, (short)0, acc, false, false);
  }

  if (K2 > 0) {
    __syncthreads();                 // everyone done reading A1 panel
    if (threadIdx.x < 32) {
      tdm_load_panel(A2 + (size_t)tM * 16 * K2, (unsigned int)(uintptr_t)ldsA,
                     (unsigned int)(16 * K2) >> 1);
      __builtin_amdgcn_s_wait_tensorcnt(0);
    }
    __syncthreads();
    for (int k = 0; k < K2; k += 32) {
      v16bf a = load_a_frag(ldsA, K2, mrow, half, k);
      v16bf b = load_b_frag(B2, N, k, lane, n0);
      acc = __builtin_amdgcn_wmma_f32_16x16x32_bf16(false, a, false, b, (short)0, acc, false, false);
    }
  }

  int n = n0 + mrow;                 // output column for this lane (all 8 rows share it)
  float bias = 0.0f;
  if (bias1) bias += bias1[n];
  if (bias2) bias += bias2[n];
#pragma unroll
  for (int r = 0; r < 8; ++r)
    C[(size_t)(tM * 16 + half * 8 + r) * N + n] = acc[r] + bias;
}

// ---------------- weight convert / transpose ----------------
__global__ void k_convT(const float* __restrict__ src, __bf16* __restrict__ dst,
                        int Nrows, int Kcols) {  // src: Nrows x Kcols -> dst: Kcols x Nrows
  size_t i = (size_t)blockIdx.x * blockDim.x + threadIdx.x;
  size_t total = (size_t)Nrows * Kcols;
  if (i >= total) return;
  int n = (int)(i / Kcols), k = (int)(i % Kcols);
  dst[(size_t)k * Nrows + n] = (__bf16)src[i];
}

__global__ void k_conv(const float* __restrict__ src, __bf16* __restrict__ dst, size_t n) {
  size_t i = (size_t)blockIdx.x * blockDim.x + threadIdx.x;
  if (i < n) dst[i] = (__bf16)src[i];
}

__global__ void k_zero(float* __restrict__ p, size_t nWords) {
  size_t i = (size_t)blockIdx.x * blockDim.x + threadIdx.x;
  if (i < nWords) p[i] = 0.0f;
}

// ---------------- prenet: x -> relu -> relu -> xabuf[:,0:256] (bf16) ----------------
__global__ void k_prenet(const float* __restrict__ inputs, int step,
                         const float* __restrict__ pW1, const float* __restrict__ pb1,
                         const float* __restrict__ pW2, const float* __restrict__ pb2,
                         __bf16* __restrict__ xabuf) {
  int b = blockIdx.x, t = threadIdx.x;  // 256 threads
  __shared__ float xs[MEL];
  __shared__ float p1[PRE1];
  if (t < MEL)
    xs[t] = (step == 0) ? 0.0f
                        : inputs[((size_t)b * T_DEC + (2 * step - 1)) * MEL + t];
  __syncthreads();
  float s = pb1[t];
  const float* w1 = pW1 + (size_t)t * MEL;
  for (int j = 0; j < MEL; ++j) s += w1[j] * xs[j];
  p1[t] = fmaxf(s, 0.0f);
  __syncthreads();
  float s2 = pb2[t];
  const float* w2 = pW2 + (size_t)t * PRE1;
  for (int j = 0; j < PRE1; ++j) s2 += w2[j] * p1[j];
  xabuf[(size_t)b * XA_K + t] = (__bf16)fmaxf(s2, 0.0f);
}

// ---------------- LSTM pointwise ----------------
__global__ void k_lstm_pw(const float* __restrict__ gates,
                          float* __restrict__ c, float* __restrict__ h,
                          __bf16* __restrict__ hbf, __bf16* __restrict__ hbf2, int hbf2Stride) {
  int idx = blockIdx.x * blockDim.x + threadIdx.x;  // 32*1024
  int b = idx >> 10, j = idx & 1023;
  const float* g = gates + (size_t)b * GATES;
  float gi = g[j], gf = g[1024 + j], gg = g[2048 + j], go = g[3072 + j];
  float cn = sigf(gf) * c[idx] + sigf(gi) * tanhf(gg);
  float hn = sigf(go) * tanhf(cn);
  c[idx] = cn;
  h[idx] = hn;
  __bf16 hb = (__bf16)hn;
  hbf[idx] = hb;
  if (hbf2) hbf2[(size_t)b * hbf2Stride + j] = hb;
}

// ---------------- location-sensitive attention (one block per batch row) ----------------
__global__ void k_attention(const float* __restrict__ ah, const float* __restrict__ Wq,
                            const float* __restrict__ loc_w, const float* __restrict__ Wl,
                            const float* __restrict__ vvec, const float* __restrict__ pm,
                            const int* __restrict__ mlen, const float* __restrict__ enc,
                            float* __restrict__ aw, float* __restrict__ ctx,
                            __bf16* __restrict__ xabuf, __bf16* __restrict__ dxbuf,
                            float* __restrict__ scores_out, int step) {
  int b = blockIdx.x, t = threadIdx.x;  // 256 threads, t == encoder position
  const int pad = (KSZ - 1) / 2;
  __shared__ float awp[T_ENC + KSZ - 1];
  __shared__ float pq[ADIM];
  __shared__ float es[T_ENC];
  __shared__ float red[256];

  awp[pad + t] = aw[(size_t)b * T_ENC + t];
  if (t < pad) { awp[t] = 0.0f; awp[pad + T_ENC + t] = 0.0f; }
  if (t < ADIM) {
    float s = 0.0f;
    const float* ahb = ah + (size_t)b * ARNN;
    const float* w = Wq + (size_t)t * ARNN;
    for (int j = 0; j < ARNN; ++j) s += ahb[j] * w[j];
    pq[t] = s;
  }
  __syncthreads();

  // location conv: 32 filters x 31 taps at position t
  float lf[FILT];
#pragma unroll
  for (int f = 0; f < FILT; ++f) {
    float s = 0.0f;
    const float* lw = loc_w + f * KSZ;
#pragma unroll
    for (int k = 0; k < KSZ; ++k) s += lw[k] * awp[t + k];
    lf[f] = s;
  }
  const float* pmt = pm + ((size_t)b * T_ENC + t) * ADIM;
  float e = 0.0f;
  for (int a = 0; a < ADIM; ++a) {
    float s = pq[a] + pmt[a];
    const float* wl = Wl + a * FILT;
#pragma unroll
    for (int f = 0; f < FILT; ++f) s += wl[f] * lf[f];
    e += vvec[a] * tanhf(s);
  }
  if (t >= mlen[b]) e = -1e9f;

  // masked softmax over 256 positions
  red[t] = e; __syncthreads();
  for (int o = 128; o > 0; o >>= 1) { if (t < o) red[t] = fmaxf(red[t], red[t + o]); __syncthreads(); }
  float mx = red[0]; __syncthreads();
  float ex = __expf(e - mx);
  red[t] = ex; __syncthreads();
  for (int o = 128; o > 0; o >>= 1) { if (t < o) red[t] += red[t + o]; __syncthreads(); }
  float al = ex / red[0];
  es[t] = al;
  aw[(size_t)b * T_ENC + t] = al;
  scores_out[((size_t)b * NSTEPS + step) * T_ENC + t] = al;
  __syncthreads();

  // ctx[d] = sum_t align[t] * enc[b,t,d], 512 dims, 2 per thread
#pragma unroll
  for (int dd = 0; dd < 2; ++dd) {
    int d = t + dd * 256;
    float s = 0.0f;
    const float* eb = enc + (size_t)b * T_ENC * D_ENC + d;
    for (int tt = 0; tt < T_ENC; ++tt) s += es[tt] * eb[(size_t)tt * D_ENC];
    ctx[(size_t)b * D_ENC + d] = s;
    __bf16 sb = (__bf16)s;
    xabuf[(size_t)b * XA_K + PRE2 + d] = sb;  // for next step's attention-LSTM input
    dxbuf[(size_t)b * DX_K + ARNN + d] = sb;  // for this step's decoder-LSTM input
  }
}

// ---------------- mel / stop projections ----------------
__global__ void k_proj(const float* __restrict__ dh, const float* __restrict__ ctx,
                       const float* __restrict__ Wmel, const float* __restrict__ bmel,
                       const float* __restrict__ Wstop, const float* __restrict__ bstop,
                       float* __restrict__ mel_out, float* __restrict__ stop_out, int step) {
  int b = blockIdx.x, t = threadIdx.x;  // 256 threads
  __shared__ float proj[DX_K];          // [dh | ctx] = 1536
  for (int j = t; j < DX_K; j += blockDim.x)
    proj[j] = (j < DRNN) ? dh[(size_t)b * DRNN + j] : ctx[(size_t)b * D_ENC + (j - DRNN)];
  __syncthreads();
  if (t < MEL * RR) {
    float s = bmel[t];
    const float* w = Wmel + (size_t)t * DX_K;
    for (int j = 0; j < DX_K; ++j) s += w[j] * proj[j];
    int r = t / MEL, m2 = t % MEL;
    mel_out[((size_t)b * T_DEC + step * RR + r) * MEL + m2] = s;
  } else if (t == MEL * RR) {
    float s = bstop[0];
    for (int j = 0; j < DX_K; ++j) s += Wstop[j] * proj[j];
    float st = sigf(s);
    stop_out[(size_t)b * T_DEC + step * RR]     = st;
    stop_out[(size_t)b * T_DEC + step * RR + 1] = st;
  }
}

// ---------------- host side ----------------
static inline size_t alignUp(size_t x) { return (x + 255) & ~(size_t)255; }

extern "C" void kernel_launch(void* const* d_in, const int* in_sizes, int n_in,
                              void* d_out, int out_size, void* d_ws, size_t ws_size,
                              hipStream_t stream) {
  (void)in_sizes; (void)n_in; (void)out_size; (void)ws_size;
  const float* enc    = (const float*)d_in[0];
  const float* inputs = (const float*)d_in[1];
  const int*   mlen   = (const int*)  d_in[2];
  const float* pW1 = (const float*)d_in[3],  *pb1 = (const float*)d_in[4];
  const float* pW2 = (const float*)d_in[5],  *pb2 = (const float*)d_in[6];
  const float* aWih = (const float*)d_in[7], *aWhh = (const float*)d_in[8];
  const float* abih = (const float*)d_in[9], *abhh = (const float*)d_in[10];
  const float* Wq = (const float*)d_in[11],  *loc_w = (const float*)d_in[12];
  const float* Wl = (const float*)d_in[13],  *vvec  = (const float*)d_in[14];
  const float* Wmem = (const float*)d_in[15];
  const float* dWih = (const float*)d_in[16], *dWhh = (const float*)d_in[17];
  const float* dbih = (const float*)d_in[18], *dbhh = (const float*)d_in[19];
  const float* Wmel = (const float*)d_in[20], *bmel = (const float*)d_in[21];
  const float* Wstop = (const float*)d_in[22], *bstop = (const float*)d_in[23];

  // output layout: mel (32*500*80) | stop (32*500) | scores (32*250*256)
  float* out      = (float*)d_out;
  float* mel_out  = out;
  float* stop_out = out + (size_t)BATCH * T_DEC * MEL;
  float* scores   = stop_out + (size_t)BATCH * T_DEC;

  // workspace layout
  char* ws = (char*)d_ws;
  size_t off = 0;
  float*  pm     = (float*)(ws + off); off = alignUp(off + (size_t)BATCH * T_ENC * ADIM * 4);
  __bf16* encbf  = (__bf16*)(ws + off); off = alignUp(off + (size_t)BATCH * T_ENC * D_ENC * 2);
  __bf16* aWihT  = (__bf16*)(ws + off); off = alignUp(off + (size_t)XA_K * GATES * 2);
  __bf16* aWhhT  = (__bf16*)(ws + off); off = alignUp(off + (size_t)ARNN * GATES * 2);
  __bf16* dWihT  = (__bf16*)(ws + off); off = alignUp(off + (size_t)DX_K * GATES * 2);
  __bf16* dWhhT  = (__bf16*)(ws + off); off = alignUp(off + (size_t)DRNN * GATES * 2);
  __bf16* WmemT  = (__bf16*)(ws + off); off = alignUp(off + (size_t)D_ENC * ADIM * 2);
  float*  gates  = (float*)(ws + off); off = alignUp(off + (size_t)BATCH * GATES * 4);
  // --- contiguous zero-initialized recurrent state region ---
  size_t stateOff = off;
  float*  ah   = (float*)(ws + off); off += (size_t)BATCH * ARNN * 4;
  float*  ac   = (float*)(ws + off); off += (size_t)BATCH * ARNN * 4;
  float*  dh   = (float*)(ws + off); off += (size_t)BATCH * DRNN * 4;
  float*  dc   = (float*)(ws + off); off += (size_t)BATCH * DRNN * 4;
  float*  ctx  = (float*)(ws + off); off += (size_t)BATCH * D_ENC * 4;
  float*  aw   = (float*)(ws + off); off += (size_t)BATCH * T_ENC * 4;
  __bf16* ahbf = (__bf16*)(ws + off); off += (size_t)BATCH * ARNN * 2;
  __bf16* dhbf = (__bf16*)(ws + off); off += (size_t)BATCH * DRNN * 2;
  __bf16* xabuf = (__bf16*)(ws + off); off += (size_t)BATCH * XA_K * 2;
  __bf16* dxbuf = (__bf16*)(ws + off); off += (size_t)BATCH * DX_K * 2;
  size_t stateWords = (off - stateOff + 3) / 4;

  // ---- setup: zero state, convert weights to bf16 (K-major), processed_memory GEMM ----
  k_zero<<<(unsigned)((stateWords + 255) / 256), 256, 0, stream>>>((float*)(ws + stateOff), stateWords);
  {
    size_t n = (size_t)BATCH * T_ENC * D_ENC;
    k_conv<<<(unsigned)((n + 255) / 256), 256, 0, stream>>>(enc, encbf, n);
  }
  auto convT = [&](const float* src, __bf16* dst, int N, int K) {
    size_t n = (size_t)N * K;
    k_convT<<<(unsigned)((n + 255) / 256), 256, 0, stream>>>(src, dst, N, K);
  };
  convT(aWih, aWihT, GATES, XA_K);
  convT(aWhh, aWhhT, GATES, ARNN);
  convT(dWih, dWihT, GATES, DX_K);
  convT(dWhh, dWhhT, GATES, DRNN);
  convT(Wmem, WmemT, ADIM, D_ENC);

  // processed_memory: (B*T_ENC x D_ENC) * (D_ENC x ADIM); tilesN=8 -> 1 block per M-tile
  k_wmma_gemm<<<512, 256, 16 * D_ENC * 2, stream>>>(encbf, WmemT, D_ENC,
                                                    nullptr, nullptr, 0, nullptr, nullptr,
                                                    pm, BATCH * T_ENC, ADIM);

  const unsigned shmAttn = 16 * ((ARNN > XA_K) ? ARNN : XA_K) * 2;  // 32 KB
  const unsigned shmDec  = 16 * ((DX_K > DRNN) ? DX_K : DRNN) * 2;  // 48 KB

  // ---- sequential decode ----
  for (int s = 0; s < NSTEPS; ++s) {
    k_prenet<<<BATCH, 256, 0, stream>>>(inputs, s, pW1, pb1, pW2, pb2, xabuf);
    // attention LSTM gates: [prenet|ctx](768) * aWihT + ah(1024) * aWhhT + biases
    k_wmma_gemm<<<64, 256, shmAttn, stream>>>(xabuf, aWihT, XA_K, ahbf, aWhhT, ARNN,
                                              abih, abhh, gates, BATCH, GATES);
    k_lstm_pw<<<(BATCH * ARNN) / 256, 256, 0, stream>>>(gates, ac, ah, ahbf, dxbuf, DX_K);
    k_attention<<<BATCH, 256, 0, stream>>>(ah, Wq, loc_w, Wl, vvec, pm, mlen, enc,
                                           aw, ctx, xabuf, dxbuf, scores, s);
    // decoder LSTM gates: [ah|ctx](1536) * dWihT + dh(1024) * dWhhT + biases
    k_wmma_gemm<<<64, 256, shmDec, stream>>>(dxbuf, dWihT, DX_K, dhbf, dWhhT, DRNN,
                                             dbih, dbhh, gates, BATCH, GATES);
    k_lstm_pw<<<(BATCH * DRNN) / 256, 256, 0, stream>>>(gates, dc, dh, dhbf, nullptr, 0);
    k_proj<<<BATCH, 256, 0, stream>>>(dh, ctx, Wmel, bmel, Wstop, bstop,
                                      mel_out, stop_out, s);
  }
}
